// TxaFilterEnsemble_8091718385911
// MI455X (gfx1250) — matrix operand order repaired and checked by reading, not compiled, not run on hardware
//
#include <hip/hip_runtime.h>

typedef __attribute__((ext_vector_type(16))) _Float16 v16h;
typedef __attribute__((ext_vector_type(8)))  float    v8f;

// Problem constants
#define BATCH 64
#define TLEN  8192
#define CCN   128        // C*2 interleaved real/imag channels
#define KTAPS 129
#define OLEN  8064

// Tiling
#define CCB   32         // channels per workgroup
#define OTILE 256        // outputs per GEMM tile: o = o0 + 16*m + n, m,n in [0,16)
#define NTILE 8          // o-tiles per workgroup
#define WPADL 16         // leading zero pad so B[k][n]=w[k-n] reads are branch-free
#define WSTR  176        // 16 zero pad + 129 taps + band/step pad, even stride
#define TPAD  408        // x stage stride (needs >= 400)
#define OPAD  33         // out stage stride (bank-conflict pad)

__global__ __launch_bounds__(256)
void txa_filter_wmma(const float* __restrict__ x,
                     const float* __restrict__ w,
                     float* __restrict__ out)
{
    __shared__ _Float16 wlds[CCB * WSTR];                          // 11264 B
    __shared__ __align__(16) unsigned char ubuf[OTILE * OPAD * 4]; // 33792 B
    _Float16* xz = (_Float16*)ubuf;   // [CCB][TPAD] f16 x stage (26112 B)
    float*    oz = (float*)ubuf;      // [OTILE][OPAD] f32 out stage

    const int tid  = threadIdx.x;
    const int lane = tid & 31;
    const int wave = tid >> 5;

    const int osuper = blockIdx.x;          // 0..3
    const int ccbase = blockIdx.y * CCB;    // 0..96
    const int b      = blockIdx.z;          // 0..63

    // Stage per-channel taps (f32 -> f16): 16 leading zeros, taps at [16,145),
    // zeros to WSTR. Index 16 + (k - n) is then always in [1,176) -> no branch.
    for (int i = tid; i < CCB * WSTR; i += 256) {
        int cl = i / WSTR, kk = (i % WSTR) - WPADL;
        float v = (kk >= 0 && kk < KTAPS) ? w[(size_t)(ccbase + cl) * KTAPS + kk] : 0.0f;
        wlds[i] = (_Float16)v;
    }

    const int m = lane & 15;   // A: row M; B: col N; C/D: col N
    const int g = lane >> 4;   // half-wave group per 16-bit WMMA layouts

    for (int ot = 0; ot < NTILE; ++ot) {
        const int o0 = (osuper * NTILE + ot) * OTILE;

        __syncthreads();  // previous tile's oz reads done before reuse as xz

        // Stage x tile: 400 times x 32 channels, coalesced (lanes = channels).
        for (int i = tid; i < 400 * CCB; i += 256) {
            int tl = i / CCB, cl = i % CCB;
            int t = o0 + tl;
            float v = (t < TLEN) ? x[((size_t)b * TLEN + t) * CCN + ccbase + cl] : 0.0f;
            xz[cl * TPAD + tl] = (_Float16)v;
        }

        // Prefetch next tile's x rows toward L2 while this tile computes.
        if (ot + 1 < NTILE) {
            const int on = o0 + OTILE;
            for (int i = tid; i < 400; i += 256) {
                int t = on + i;
                if (t < TLEN)
                    __builtin_prefetch(&x[((size_t)b * TLEN + t) * CCN + ccbase], 0, 1);
            }
        }
        __syncthreads();

        // Each wave computes 4 channels; per channel one 16x16 tile (256 outputs)
        // via a K=160 banded-Toeplitz reduction = 5x v_wmma_f32_16x16x32_f16.
        v8f acc[4];
        #pragma unroll
        for (int ci = 0; ci < 4; ++ci) {
            const int cl = wave * 4 + ci;
            v8f c = {};
            #pragma unroll
            for (int ks = 0; ks < 5; ++ks) {
                const int kbase = ks * 32;
                // A[m][k] = x[o0 + 16*m + kbase + k]  (Hankel, rows stride 16)
                // A k order used here (contiguous, same as B): half h -> k = 16*g + h
                v16h a;
                #pragma unroll
                for (int h = 0; h < 16; ++h) {
                    int k = 16 * g + h;
                    a[h] = xz[cl * TPAD + 16 * m + kbase + k];
                }
                // B[k][n] = w[kbase + k - n] via zero-padded wlds, branch-free
                // B: lane n = lane&15 (== m), k order as A: half h -> k = 16*g + h
                v16h bm;
                #pragma unroll
                for (int h = 0; h < 16; ++h) {
                    int idx = WPADL + kbase + 16 * g + h - m;   // in [1, 176)
                    bm[h] = wlds[cl * WSTR + idx];
                }
                c = __builtin_amdgcn_wmma_f32_16x16x32_f16(false, a, false, bm,
                                                           (short)0, c, false, false);
            }
            acc[ci] = c;
        }

        __syncthreads();  // all xz reads done before reuse as oz

        // C/D layout: lane N = lane&15, VGPR v holds M = v + 8*g. o_local = 16*M + N.
        #pragma unroll
        for (int ci = 0; ci < 4; ++ci) {
            const int cl = wave * 4 + ci;
            #pragma unroll
            for (int v = 0; v < 8; ++v) {
                int M = v + 8 * g;
                oz[(16 * M + m) * OPAD + cl] = acc[ci][v];
            }
        }
        __syncthreads();

        // Coalesced global store (lanes = channels), masked at O tail.
        for (int i = tid; i < OTILE * CCB; i += 256) {
            int ol = i / CCB, cl = i % CCB;
            int o = o0 + ol;
            if (o < OLEN)
                out[((size_t)b * OLEN + o) * CCN + ccbase + cl] = oz[ol * OPAD + cl];
        }
    }
}

extern "C" void kernel_launch(void* const* d_in, const int* in_sizes, int n_in,
                              void* d_out, int out_size, void* d_ws, size_t ws_size,
                              hipStream_t stream) {
    const float* x = (const float*)d_in[0];   // [64, 8192, 64, 2] f32
    const float* w = (const float*)d_in[1];   // [64, 2, 129] f32
    float* out = (float*)d_out;               // [64, 8064, 64, 2] f32
    dim3 grid(4, CCN / CCB, BATCH);           // (o-supertile, channel-block, batch)
    dim3 block(256);
    txa_filter_wmma<<<grid, block, 0, stream>>>(x, w, out);
    (void)in_sizes; (void)n_in; (void)out_size; (void)d_ws; (void)ws_size;
}
